// VectorQuantizer_68685116998172
// MI455X (gfx1250) — matrix operand order repaired and checked by reading, not compiled, not run on hardware
//
#include <hip/hip_runtime.h>
#include <hip/hip_bf16.h>

typedef __attribute__((ext_vector_type(16))) __bf16 v16bf;
typedef __attribute__((ext_vector_type(8)))  __bf16 v8bf;
typedef __attribute__((ext_vector_type(8)))  float  v8f;

#define BN   4096
#define EN   16384
#define DIMN 4096

// ---------------- init: reset argmin slots, counts, loss ----------------
__global__ void vq_init(unsigned long long* __restrict__ packed,
                        unsigned int* __restrict__ counts,
                        float* __restrict__ loss) {
    int t = blockIdx.x * 256 + threadIdx.x;
    if (t < EN)  counts[t] = 0u;
    if (t < BN)  packed[t] = ~0ULL;
    if (t == 0)  *loss = 0.0f;
}

// ---------------- zero the one-hot encodings (256 MB) ----------------
__global__ void vq_zero_enc(float4* __restrict__ enc) {
    size_t t = (size_t)blockIdx.x * 256 + threadIdx.x;   // 16,777,216 threads
    enc[t] = make_float4(0.f, 0.f, 0.f, 0.f);
}

// ---------------- f32 -> bf16 copy (packed 16B stores) + row norms ----------------
__global__ void vq_convert_norm(const float* __restrict__ src,
                                __bf16* __restrict__ dst,
                                float* __restrict__ norms) {
    const int row = blockIdx.x;
    const float* s = src + (size_t)row * DIMN;
    __bf16* d = dst + (size_t)row * DIMN;
    float acc = 0.f;
    // each thread converts 8 contiguous floats per step -> one v8bf (16B) store
    for (int base = 0; base < DIMN; base += 256 * 8) {
        const int i = base + threadIdx.x * 8;
        float4 f0 = *(const float4*)(s + i);
        float4 f1 = *(const float4*)(s + i + 4);
        acc += f0.x * f0.x + f0.y * f0.y + f0.z * f0.z + f0.w * f0.w;
        acc += f1.x * f1.x + f1.y * f1.y + f1.z * f1.z + f1.w * f1.w;
        v8bf o;
        o[0] = (__bf16)f0.x; o[1] = (__bf16)f0.y; o[2] = (__bf16)f0.z; o[3] = (__bf16)f0.w;
        o[4] = (__bf16)f1.x; o[5] = (__bf16)f1.y; o[6] = (__bf16)f1.z; o[7] = (__bf16)f1.w;
        *(v8bf*)(d + i) = o;
    }
    __shared__ float red[256];
    red[threadIdx.x] = acc;
    __syncthreads();
    for (int o = 128; o > 0; o >>= 1) {
        if (threadIdx.x < o) red[threadIdx.x] += red[threadIdx.x + o];
        __syncthreads();
    }
    if (threadIdx.x == 0) norms[row] = red[0];
}

// A-fragment loader: 16-bit A 16x32 layout (two 16B chunks per lane)
__device__ __forceinline__ v16bf load_afrag(const __bf16* p) {
    v8bf lo = *(const v8bf*)p;          // K = half*8 + [0..7]
    v8bf hi = *(const v8bf*)(p + 16);   // K = 16 + half*8 + [0..7]
    return __builtin_shufflevector(lo, hi,
               0, 1, 2, 3, 4, 5, 6, 7, 8, 9, 10, 11, 12, 13, 14, 15);
}

// Epilogue for one 16-row M strip against two 16-col N strips.
__device__ __forceinline__ void argmin_emit(v8f cA, v8f cB, int mbase,
                                            int half, int lm,
                                            float enA, float enB,
                                            unsigned int iA, unsigned int iB,
                                            const float* __restrict__ xn,
                                            unsigned long long* __restrict__ packed) {
    #pragma unroll
    for (int r = 0; r < 8; ++r) {
        const int m = mbase + half * 8 + r;        // C layout: lanes16-31 -> M+8
        const float xnm = xn[m];
        float d0 = xnm + enA - 2.0f * cA[r];
        float d1 = xnm + enB - 2.0f * cB[r];
        float dm; unsigned int idx;
        if (d1 < d0) { dm = d1; idx = iB; } else { dm = d0; idx = iA; }
        unsigned int kb = __float_as_uint(dm);     // monotonic float -> uint key
        kb = kb ^ (unsigned int)(((int)kb >> 31) | 0x80000000);
        unsigned long long p = ((unsigned long long)kb << 32) | (unsigned long long)idx;
        #pragma unroll
        for (int off = 8; off >= 1; off >>= 1) {
            unsigned long long o = __shfl_xor(p, off, 16);
            if (o < p) p = o;
        }
        if (lm == 0) atomicMin(&packed[m], p);
    }
}

// ---------------- main GEMM + fused distance + argmin ----------------
// Block: 256 threads = 8 waves; block tile = 128(M) x 64(N).
// Wave w: M strip = (w>>1)*32, N strip = (w&1)*32 -> 32x32 register tile
// = 4 x v_wmma_f32_16x16x32_bf16 per K-step (2 A-frags x 2 B-frags).
// Operands (32 MB x-bf16 + 128 MB e-bf16) are L2-resident; the 8-wave
// blocks + deep unroll cover L2 latency without prefetch (prefetch asm
// forced WMMA->VALU hazard NOPs, so it is deliberately absent).
__global__ void __launch_bounds__(256)
vq_gemm_argmin(const __bf16* __restrict__ xb,
               const __bf16* __restrict__ eb,
               const float* __restrict__ xn,
               const float* __restrict__ en,
               unsigned long long* __restrict__ packed) {
    const int lane = threadIdx.x & 31;
    const int wave = threadIdx.x >> 5;
    const int lm   = lane & 15;
    const int half = lane >> 4;

    const int m0 = blockIdx.y * 128 + (wave >> 1) * 32;
    const int n0 = blockIdx.x * 64  + (wave & 1) * 32;

    const __bf16* pa0 = xb + (size_t)(m0 + lm) * DIMN + half * 8;
    const __bf16* pa1 = pa0 + (size_t)16 * DIMN;
    const __bf16* pb0 = eb + (size_t)(n0 + lm) * DIMN + half * 8;
    const __bf16* pb1 = pb0 + (size_t)16 * DIMN;

    v8f c00 = {}; v8f c01 = {};
    v8f c10 = {}; v8f c11 = {};

    for (int k = 0; k < DIMN; k += 32) {
        v16bf a0 = load_afrag(pa0 + k);
        v16bf a1 = load_afrag(pa1 + k);
        v16bf b0 = load_afrag(pb0 + k);   // K = k + half*8 + [0..7], k + 16 + half*8 + [0..7]
        v16bf b1 = load_afrag(pb1 + k);

        c00 = __builtin_amdgcn_wmma_f32_16x16x32_bf16(
                  false, a0, false, b0, (short)0, c00, false, false);
        c01 = __builtin_amdgcn_wmma_f32_16x16x32_bf16(
                  false, a0, false, b1, (short)0, c01, false, false);
        c10 = __builtin_amdgcn_wmma_f32_16x16x32_bf16(
                  false, a1, false, b0, (short)0, c10, false, false);
        c11 = __builtin_amdgcn_wmma_f32_16x16x32_bf16(
                  false, a1, false, b1, (short)0, c11, false, false);
    }

    const float en0 = en[n0 + lm];
    const float en1 = en[n0 + 16 + lm];
    const unsigned int i0 = (unsigned)(n0 + lm);
    const unsigned int i1 = (unsigned)(n0 + 16 + lm);

    argmin_emit(c00, c01, m0,      half, lm, en0, en1, i0, i1, xn, packed);
    argmin_emit(c10, c11, m0 + 16, half, lm, en0, en1, i0, i1, xn, packed);
}

// ---------------- extract indices, one-hot, usage counts ----------------
__global__ void vq_scatter(const unsigned long long* __restrict__ packed,
                           float* __restrict__ enc,
                           unsigned int* __restrict__ counts,
                           unsigned int* __restrict__ idxArr) {
    int b = blockIdx.x * 256 + threadIdx.x;
    if (b < BN) {
        unsigned int idx = (unsigned int)(packed[b] & 0xFFFFFFFFULL);
        idxArr[b] = idx;
        enc[(size_t)b * EN + idx] = 1.0f;
        atomicAdd(&counts[idx], 1u);
    }
}

// ---------------- quantized output + loss (float4 per thread) ----------------
__global__ void vq_quant_loss(const float* __restrict__ inputs,
                              const float* __restrict__ embed,
                              const unsigned int* __restrict__ idxArr,
                              float* __restrict__ quant,
                              float* __restrict__ loss) {
    size_t t4 = ((size_t)blockIdx.x * 256 + threadIdx.x) * 4;  // B*DIM/4 threads
    int b = (int)(t4 >> 12);            // / 4096 (all 4 elems same row)
    int d = (int)(t4 & (DIMN - 1));
    float4 q = *(const float4*)(embed + (size_t)idxArr[b] * DIMN + d);
    float4 x = *(const float4*)(inputs + t4);
    *(float4*)(quant + t4) = q;
    float dx = q.x - x.x, dy = q.y - x.y, dz = q.z - x.z, dw = q.w - x.w;
    float v = dx * dx + dy * dy + dz * dz + dw * dw;

    __shared__ float red[256];
    red[threadIdx.x] = v;
    __syncthreads();
    for (int o = 128; o > 0; o >>= 1) {
        if (threadIdx.x < o) red[threadIdx.x] += red[threadIdx.x + o];
        __syncthreads();
    }
    if (threadIdx.x == 0)
        atomicAdd(loss, red[0] * (1.25f / (float)((size_t)BN * DIMN)));
}

// ---------------- perplexity ----------------
__global__ void vq_perplexity(const unsigned int* __restrict__ counts,
                              float* __restrict__ out) {
    float s = 0.f;
    for (int e = threadIdx.x; e < EN; e += 256) {
        float p = (float)counts[e] * (1.0f / (float)BN);
        s += -p * logf(p + 1e-10f);
    }
    __shared__ float red[256];
    red[threadIdx.x] = s;
    __syncthreads();
    for (int o = 128; o > 0; o >>= 1) {
        if (threadIdx.x < o) red[threadIdx.x] += red[threadIdx.x + o];
        __syncthreads();
    }
    if (threadIdx.x == 0) *out = expf(red[0]);
}

extern "C" void kernel_launch(void* const* d_in, const int* in_sizes, int n_in,
                              void* d_out, int out_size, void* d_ws, size_t ws_size,
                              hipStream_t stream) {
    const float* inputs = (const float*)d_in[0];   // [4096, 64,4,4,4] f32
    const float* embed  = (const float*)d_in[1];   // [16384, 4096]   f32

    // d_out layout: loss(1) | quantized(16,777,216) | perplexity(1) | encodings(67,108,864)
    float* out   = (float*)d_out;
    float* loss  = out;
    float* quant = out + 1;
    float* ppl   = out + 1 + (size_t)BN * DIMN;
    float* enc   = ppl + 1;

    // workspace carve-up (256-aligned regions)
    char* ws = (char*)d_ws;
    __bf16* xb = (__bf16*)ws;                                   //  32 MB
    __bf16* eb = (__bf16*)(ws + (size_t)33554432);              // 128 MB
    float*  xn = (float*) (ws + (size_t)33554432 + 134217728);  //  16 KB
    float*  en = xn + BN;                                       //  64 KB
    unsigned long long* packed = (unsigned long long*)(en + EN);
    unsigned int* counts = (unsigned int*)(packed + BN);
    unsigned int* idxArr = counts + EN;

    vq_init<<<64, 256, 0, stream>>>(packed, counts, loss);
    vq_zero_enc<<<65536, 256, 0, stream>>>((float4*)enc);
    vq_convert_norm<<<BN, 256, 0, stream>>>(inputs, xb, xn);
    vq_convert_norm<<<EN, 256, 0, stream>>>(embed, eb, en);

    dim3 ggrid(EN / 64, BN / 128);   // 256 x 32 blocks
    vq_gemm_argmin<<<ggrid, 256, 0, stream>>>(xb, eb, xn, en, packed);

    vq_scatter<<<BN / 256, 256, 0, stream>>>(packed, enc, counts, idxArr);
    vq_quant_loss<<<16384, 256, 0, stream>>>(inputs, embed, idxArr, quant, loss);
    vq_perplexity<<<1, 256, 0, stream>>>(counts, ppl);
}
